// PointAttention_37838661877830
// MI455X (gfx1250) — matrix operand order repaired and checked by reading, not compiled, not run on hardware
//
#include <hip/hip_runtime.h>
#include <hip/hip_bf16.h>

#define IN_DIM 512
#define LAYER  128
#define BATCH  4
#define NPTS   4096

typedef __attribute__((ext_vector_type(16))) __bf16        v16bf;
typedef __attribute__((ext_vector_type(8)))  float         v8f;
typedef __attribute__((ext_vector_type(4)))  unsigned int  u32x4;
typedef __attribute__((ext_vector_type(4)))  int           i32x4;
typedef __attribute__((ext_vector_type(8)))  int           i32x8;

union BF16Frag {
    v16bf          v;
    u32x4          q[2];
    unsigned int   u[8];
    unsigned short s[16];
};

// round-to-nearest-even f32 -> bf16 bits (single)
static __device__ __forceinline__ unsigned short f2bf(float f) {
    unsigned u = __float_as_uint(f);
    u += 0x7FFFu + ((u >> 16) & 1u);
    return (unsigned short)(u >> 16);
}

// pack two f32 -> one dword of 2 bf16 (lo, hi), RNE
static __device__ __forceinline__ unsigned int pack_bf2(float lo, float hi) {
    unsigned a = __float_as_uint(lo); a += 0x7FFFu + ((a >> 16) & 1u);
    unsigned b = __float_as_uint(hi); b += 0x7FFFu + ((b >> 16) & 1u);
    return (a >> 16) | (b & 0xFFFF0000u);
}

static __device__ __forceinline__ v8f wmma_bf16(const BF16Frag& a, const BF16Frag& b, v8f c) {
    return __builtin_amdgcn_wmma_f32_16x16x32_bf16(false, a.v, false, b.v,
                                                   (short)0, c, false, false);
}

// A fragment (16x32 bf16): lane row = lane%16, K run 16h..16h+15
// -> two contiguous 16-byte loads from a row-major row pointer.
static __device__ __forceinline__ void load_a_frag(BF16Frag& f, const unsigned short* p, int h) {
    f.q[0] = *(const u32x4*)(p + 16 * h);
    f.q[1] = *(const u32x4*)(p + 16 * h + 8);
}
// B fragment (32x16 bf16): lane col = lane%16, K run 16h..16h+15
// -> two contiguous 16-byte loads from a column pointer (K-major).
static __device__ __forceinline__ void load_b_frag(BF16Frag& f, const unsigned short* p, int h) {
    f.q[0] = *(const u32x4*)(p + 16 * h);
    f.q[1] = *(const u32x4*)(p + 16 * h + 8);
}

// ---------------------------------------------------------------------------
// Tensor Data Mover: 2D f32 tile (tile_x contiguous elems x tile_y rows,
// row stride = stride_elems) from global -> LDS.  D# packing per ISA §8.
// ---------------------------------------------------------------------------
static __device__ __forceinline__ void tdm_load_2d_f32(unsigned lds_byte_off,
                                                       const void* gptr,
                                                       int tile_x, int tile_y,
                                                       int stride_elems) {
    unsigned long long ga = (unsigned long long)(size_t)gptr;
    u32x4 g0;
    g0[0] = 1u;                                              // count=1, user mode
    g0[1] = lds_byte_off;                                    // lds_addr
    g0[2] = (unsigned)(ga & 0xFFFFFFFFu);                    // global_addr[31:0]
    g0[3] = (unsigned)((ga >> 32) & 0x01FFFFFFu) | (2u << 30); // addr[56:32] | type=2
    i32x8 g1;
    g1[0] = (2 << 16);                                       // data_size = 4 bytes
    g1[1] = (int)((NPTS & 0xFFFF) << 16);                    // tensor_dim0[15:0]
    g1[2] = (int)(((NPTS >> 16) & 0xFFFF) |                  // tensor_dim0[31:16]
                  ((IN_DIM & 0xFFFF) << 16));                // tensor_dim1[15:0]
    g1[3] = (int)(((IN_DIM >> 16) & 0xFFFF) |                // tensor_dim1[31:16]
                  ((tile_x & 0xFFFF) << 16));                // tile_dim0
    g1[4] = (int)(tile_y & 0xFFFF);                          // tile_dim1 (tile_dim2=0)
    g1[5] = stride_elems;                                    // tensor_dim0_stride[31:0]
    g1[6] = 0;
    g1[7] = 0;
    i32x4 gz = {0, 0, 0, 0};
#if defined(__clang_major__) && (__clang_major__ >= 23)
    i32x8 gz8 = {0, 0, 0, 0, 0, 0, 0, 0};
    __builtin_amdgcn_tensor_load_to_lds(g0, g1, gz, gz, gz8, 0);
#else
    __builtin_amdgcn_tensor_load_to_lds(g0, g1, gz, gz, 0);
#endif
}

// ---------------------------------------------------------------------------
// Kernel 1: projection GEMM  out[oc][n] = sum_c W[oc][c]*X[c][n] + bias[oc]
// Block = 4 waves sharing one n-tile; oc-tile = blockIdx.z*4 + wave.
// The strided X tile (32c x 16n f32) is DMA'd into LDS by the TDM; A (W) rows
// load as contiguous float4s.  trans=1 stores [n][OC] (Qt/Kt), else [OC][n].
// ---------------------------------------------------------------------------
__global__ void proj_kernel(const float* __restrict__ inp,
                            const float* __restrict__ W,
                            const float* __restrict__ bias,
                            unsigned short* __restrict__ out,
                            int OC, int trans) {
    __shared__ __align__(16) float Btile[32][16];

    const int wave = threadIdx.x >> 5;
    const int lane = threadIdx.x & 31;
    const int h    = lane >> 4;
    const int lm   = lane & 15;

    const int nt  = blockIdx.x;
    const int b   = blockIdx.y;
    const int oct = blockIdx.z * 4 + wave;

    const int n0     = nt * 16;
    const int n_col  = n0 + lm;
    const int oc_row = oct * 16 + lm;
    const float* inb = inp + (size_t)b * IN_DIM * NPTS;

    v8f acc = {};
    for (int kk = 0; kk < IN_DIM; kk += 32) {
        __syncthreads();                       // previous tile fully consumed
        if (wave == 0) {
            tdm_load_2d_f32((unsigned)(size_t)(void*)&Btile[0][0],
                            inb + (size_t)kk * NPTS + n0,
                            /*tile_x=*/16, /*tile_y=*/32, /*stride=*/NPTS);
            __builtin_amdgcn_s_wait_tensorcnt(0);
        }
        __syncthreads();                       // tile visible to all waves

        BF16Frag a, bf;
        const float* wr = W + (size_t)oc_row * IN_DIM + kk;
#pragma unroll
        for (int g = 0; g < 2; ++g) {
            const float* wp = wr + g * 8 + 16 * h;   // 8 consecutive f32
#pragma unroll
            for (int j = 0; j < 4; ++j)
                a.u[g * 4 + j] = pack_bf2(wp[2 * j], wp[2 * j + 1]);
        }
#pragma unroll
        for (int v = 0; v < 8; ++v) {
            int kb = 16 * h + 2 * v;
            bf.u[v] = pack_bf2(Btile[kb][lm], Btile[kb + 1][lm]);
        }
        acc = wmma_bf16(a, bf, acc);
    }

    if (trans) {
        // 8 consecutive oc values per lane -> single 16B store
        const int oc0 = oct * 16 + 8 * h;
        u32x4 st;
#pragma unroll
        for (int j = 0; j < 4; ++j)
            st[j] = pack_bf2(acc[2 * j]     + bias[oc0 + 2 * j],
                             acc[2 * j + 1] + bias[oc0 + 2 * j + 1]);
        *(u32x4*)(out + ((size_t)b * NPTS + n_col) * OC + oc0) = st;
    } else {
#pragma unroll
        for (int r = 0; r < 8; ++r) {
            int oc = oct * 16 + r + 8 * h;
            out[((size_t)b * OC + oc) * NPTS + n_col] = f2bf(acc[r] + bias[oc]);
        }
    }
}

// ---------------------------------------------------------------------------
// Kernel 2: softmax statistics — per-row max r[n] and 1/sum(exp(S-r)).
// S tile = Qt(rows n) x Kt(cols m), K=128 -> 4 WMMA per tile; online max/sum
// with __shfl_xor reductions inside 16-lane halves (C-matrix layout).
// ---------------------------------------------------------------------------
__global__ void pass1_kernel(const unsigned short* __restrict__ Qt,
                             const unsigned short* __restrict__ Kt,
                             float* __restrict__ rmax,
                             float* __restrict__ linv) {
    const int wave = threadIdx.x >> 5;
    const int lane = threadIdx.x & 31;
    const int h    = lane >> 4;
    const int lm   = lane & 15;
    const int nt   = blockIdx.x * 4 + wave;
    const int b    = blockIdx.y;

    const unsigned short* Qb = Qt + (size_t)b * NPTS * LAYER;
    const unsigned short* Kb = Kt + (size_t)b * NPTS * LAYER;
    const int nrow = nt * 16 + lm;

    BF16Frag aq[4];
#pragma unroll
    for (int k = 0; k < 4; ++k)
        load_a_frag(aq[k], Qb + (size_t)nrow * LAYER + k * 32, h);

    float rm[8], rs[8];
#pragma unroll
    for (int r = 0; r < 8; ++r) { rm[r] = -3.0e38f; rs[r] = 0.0f; }

    for (int mt = 0; mt < NPTS / 16; ++mt) {
        const int mcol = mt * 16 + lm;
        v8f s = {};
#pragma unroll
        for (int k = 0; k < 4; ++k) {
            BF16Frag bk;
            load_b_frag(bk, Kb + (size_t)mcol * LAYER + k * 32, h);
            s = wmma_bf16(aq[k], bk, s);
        }
#pragma unroll
        for (int r = 0; r < 8; ++r) {
            float val = s[r];
            float tm = val;
            tm = fmaxf(tm, __shfl_xor(tm, 1));
            tm = fmaxf(tm, __shfl_xor(tm, 2));
            tm = fmaxf(tm, __shfl_xor(tm, 4));
            tm = fmaxf(tm, __shfl_xor(tm, 8));
            float nm = fmaxf(rm[r], tm);
            float e  = __expf(val - nm);
            e += __shfl_xor(e, 1);
            e += __shfl_xor(e, 2);
            e += __shfl_xor(e, 4);
            e += __shfl_xor(e, 8);
            rs[r] = rs[r] * __expf(rm[r] - nm) + e;
            rm[r] = nm;
        }
    }
    if (lm == 0) {                    // lane 0 -> rows 0..7, lane 16 -> 8..15
#pragma unroll
        for (int r = 0; r < 8; ++r) {
            int n = nt * 16 + r + 8 * h;
            rmax[(size_t)b * NPTS + n] = rm[r];
            linv[(size_t)b * NPTS + n] = 1.0f / rs[r];
        }
    }
}

// ---------------------------------------------------------------------------
// Kernel 3: O[:,m] = V * E', E'[n,m] = exp(S[n,m]-r[n]) / l[n]
// Block = 8 waves, one 16-col m-tile, all 512 channels.  Waves 0-1 recompute
// the 32x16 S tile (K-dim 128), exponentiate, share bf16 E' via LDS; all 8
// waves chain V(16x32) x E'(32x16) WMMA over n.  Epilogue: out = g*O + X.
// ---------------------------------------------------------------------------
__global__ void pass2_kernel(const unsigned short* __restrict__ Qt,
                             const unsigned short* __restrict__ Kt,
                             const unsigned short* __restrict__ Vb,
                             const float* __restrict__ rmax,
                             const float* __restrict__ linv,
                             const float* __restrict__ inp,
                             const float* __restrict__ gamma,
                             float* __restrict__ out) {
    __shared__ __align__(16) unsigned short Elds[16][32];   // [m][n] bf16

    const int wave = threadIdx.x >> 5;
    const int lane = threadIdx.x & 31;
    const int h    = lane >> 4;
    const int lm   = lane & 15;
    const int mt   = blockIdx.x;
    const int b    = blockIdx.y;
    const int mcol = mt * 16 + lm;
    const int cbase = wave * 64;

    const unsigned short* Qb = Qt + (size_t)b * NPTS * LAYER;
    const unsigned short* Kb = Kt + (size_t)b * NPTS * LAYER;
    const unsigned short* Vp = Vb + (size_t)b * IN_DIM * NPTS;
    const float* rb = rmax + (size_t)b * NPTS;
    const float* lb = linv + (size_t)b * NPTS;

    v8f acc[4];
#pragma unroll
    for (int i = 0; i < 4; ++i) acc[i] = (v8f){};

    BF16Frag bk[4];                      // Kt column: loop-invariant, hoisted
    if (wave < 2) {
#pragma unroll
        for (int k = 0; k < 4; ++k)
            load_b_frag(bk[k], Kb + (size_t)mcol * LAYER + k * 32, h);
    }

    for (int ch = 0; ch < NPTS / 32; ++ch) {
        const int n0 = ch * 32;
        if (wave < 2) {
            const int nbase = n0 + wave * 16;
            const int nrow  = nbase + lm;
            v8f s = {};
#pragma unroll
            for (int k = 0; k < 4; ++k) {
                BF16Frag aq;
                load_a_frag(aq, Qb + (size_t)nrow * LAYER + k * 32, h);
                s = wmma_bf16(aq, bk[k], s);
            }
#pragma unroll
            for (int r = 0; r < 8; ++r) {
                int n   = nbase + r + 8 * h;
                float e = __expf(s[r] - rb[n]) * lb[n];
                Elds[lm][wave * 16 + r + 8 * h] = f2bf(e);
            }
        }
        __syncthreads();

        BF16Frag be;                     // E' col m, K=n: 2x 16B LDS loads
        be.q[0] = *(const u32x4*)&Elds[lm][16 * h];
        be.q[1] = *(const u32x4*)&Elds[lm][16 * h + 8];

#pragma unroll
        for (int i = 0; i < 4; ++i) {
            int c = cbase + i * 16 + lm;
            BF16Frag av;
            load_a_frag(av, Vp + (size_t)c * NPTS + n0, h);
            acc[i] = wmma_bf16(av, be, acc[i]);
        }
        __syncthreads();
    }

    const float g = gamma[0];
#pragma unroll
    for (int i = 0; i < 4; ++i)
#pragma unroll
        for (int r = 0; r < 8; ++r) {
            int c = cbase + i * 16 + r + 8 * h;
            size_t idx = ((size_t)b * IN_DIM + c) * NPTS + mcol;
            out[idx] = g * acc[i][r] + inp[idx];
        }
}

// ---------------------------------------------------------------------------
extern "C" void kernel_launch(void* const* d_in, const int* in_sizes, int n_in,
                              void* d_out, int out_size, void* d_ws, size_t ws_size,
                              hipStream_t stream) {
    const float* inp   = (const float*)d_in[0];
    const float* Wq    = (const float*)d_in[1];
    const float* bq    = (const float*)d_in[2];
    const float* Wk    = (const float*)d_in[3];
    const float* bk    = (const float*)d_in[4];
    const float* Wv    = (const float*)d_in[5];
    const float* bv    = (const float*)d_in[6];
    const float* gamma = (const float*)d_in[7];

    const size_t QK_BYTES = (size_t)BATCH * NPTS * LAYER * sizeof(unsigned short);  // 4 MB
    const size_t V_BYTES  = (size_t)BATCH * IN_DIM * NPTS * sizeof(unsigned short); // 16 MB

    char* ws = (char*)d_ws;
    unsigned short* Qt = (unsigned short*)(ws);
    unsigned short* Kt = (unsigned short*)(ws + QK_BYTES);
    unsigned short* Vb = (unsigned short*)(ws + 2 * QK_BYTES);
    float*          rm = (float*)(ws + 2 * QK_BYTES + V_BYTES);
    float*          li = rm + (size_t)BATCH * NPTS;

    // projections: grid = (n-tiles, batch, oc-groups of 64)
    proj_kernel<<<dim3(NPTS / 16, BATCH, LAYER  / 64), 128, 0, stream>>>(inp, Wq, bq, Qt, LAYER, 1);
    proj_kernel<<<dim3(NPTS / 16, BATCH, LAYER  / 64), 128, 0, stream>>>(inp, Wk, bk, Kt, LAYER, 1);
    proj_kernel<<<dim3(NPTS / 16, BATCH, IN_DIM / 64), 128, 0, stream>>>(inp, Wv, bv, Vb, IN_DIM, 0);

    // softmax statistics: 256 n-tiles per batch, 4 per block
    pass1_kernel<<<dim3((NPTS / 16) / 4, BATCH), 128, 0, stream>>>(Qt, Kt, rm, li);

    // attention-apply + epilogue: one block per (m-tile, batch)
    pass2_kernel<<<dim3(NPTS / 16, BATCH), 256, 0, stream>>>(Qt, Kt, Vb, rm, li, inp, gamma,
                                                             (float*)d_out);
}
